// MultiHeadAttention_62783831933512
// MI455X (gfx1250) — matrix operand order repaired and checked by reading, not compiled, not run on hardware
//
#include <hip/hip_runtime.h>
#include <hip/hip_bf16.h>

typedef __attribute__((ext_vector_type(16))) __bf16 v16bf;
typedef __attribute__((ext_vector_type(4)))  __bf16 bf16x4;
typedef __attribute__((ext_vector_type(8)))  float  v8f;
typedef __attribute__((ext_vector_type(4)))  float  f32x4;

#define WMMA_BF16(a, b, c) \
  __builtin_amdgcn_wmma_f32_16x16x32_bf16(false, (a), false, (b), (short)0, (c), false, false)

// ---------------------------------------------------------------------------
// Fragment loaders (contiguous k order, same for A and B, wave32)
//   A (16xK of row-major [ld]): lane holds row m0+(lane&15); elements
//     e -> K=half*16+e  (one contiguous 32B run)
//   B (Kx16): lane holds col n0+(lane&15); element e -> K=(lane>>4)*16+e
//   C/D f32: lane, vgpr j -> row m0+(lane>>4)*8+j, col n0+(lane&15)
// ---------------------------------------------------------------------------

__device__ __forceinline__ v16bf load_a_f32(const float* __restrict__ A, int m0, int k0,
                                            int ld, int lane) {
  const float* rp = A + (size_t)(m0 + (lane & 15)) * ld + k0 + (lane >> 4) * 16;
  v16bf a;
#pragma unroll
  for (int e = 0; e < 8; ++e) a[e] = (__bf16)rp[e];
#pragma unroll
  for (int e = 0; e < 8; ++e) a[8 + e] = (__bf16)rp[8 + e];
  return a;
}

__device__ __forceinline__ v16bf load_a_bf16(const __bf16* __restrict__ A, int m0, int k0,
                                             int ld, int lane) {
  const __bf16* rp = A + (size_t)(m0 + (lane & 15)) * ld + k0 + (lane >> 4) * 16;
  v16bf a;
#pragma unroll
  for (int e = 0; e < 8; ++e) a[e] = rp[e];
#pragma unroll
  for (int e = 0; e < 8; ++e) a[8 + e] = rp[8 + e];
  return a;
}

// B from row-major [K x N] f32 source (weights): scattered, amortize via M-blocking
__device__ __forceinline__ v16bf load_b_f32(const float* __restrict__ W, int k0, int n0,
                                            int ld, int lane) {
  const int n = n0 + (lane & 15);
  const int kb = k0 + ((lane >> 4) << 4);
  v16bf b;
#pragma unroll
  for (int e = 0; e < 16; ++e) b[e] = (__bf16)W[(size_t)(kb + e) * ld + n];
  return b;
}

// B from row-major [K x N] bf16 source
__device__ __forceinline__ v16bf load_b_bf16(const __bf16* __restrict__ W, int k0, int n0,
                                             int ld, int lane) {
  const int n = n0 + (lane & 15);
  const int kb = k0 + ((lane >> 4) << 4);
  v16bf b;
#pragma unroll
  for (int e = 0; e < 16; ++e) b[e] = W[(size_t)(kb + e) * ld + n];
  return b;
}

// B as transpose of row-major [N x K] bf16 (Q @ K^T): contiguous 32B per lane
__device__ __forceinline__ v16bf load_bT_bf16(const __bf16* __restrict__ M, int n0, int k0,
                                              int ld, int lane) {
  const __bf16* rp = M + (size_t)(n0 + (lane & 15)) * ld + k0 + ((lane >> 4) << 4);
  v16bf b;
#pragma unroll
  for (int e = 0; e < 16; ++e) b[e] = rp[e];
  return b;
}

// ---------------------------------------------------------------------------
// Kernel 1: qh/kh/vh = X @ W + b -> bf16 [4096 x 64]
// Wave computes a 64x16 strip: 1 scattered B-frag shared by 4 WMMAs per K-step.
// grid = (64 m-quads, 4 n-tiles, 3 matrices), block = 32
// ---------------------------------------------------------------------------
__global__ __launch_bounds__(32) void proj_kernel(
    const float* __restrict__ q, const float* __restrict__ k, const float* __restrict__ v,
    const float* __restrict__ Wq, const float* __restrict__ Wk, const float* __restrict__ Wv,
    const float* __restrict__ bq, const float* __restrict__ bk, const float* __restrict__ bv,
    __bf16* __restrict__ qh, __bf16* __restrict__ kh, __bf16* __restrict__ vh) {
  const int lane = threadIdx.x & 31;
  const int which = blockIdx.z;
  const float* A    = which == 0 ? q  : (which == 1 ? k  : v);
  const float* W    = which == 0 ? Wq : (which == 1 ? Wk : Wv);
  const float* bias = which == 0 ? bq : (which == 1 ? bk : bv);
  __bf16* Out       = which == 0 ? qh : (which == 1 ? kh : vh);

  const int m0 = blockIdx.x * 64;
  const int n0 = blockIdx.y * 16;
  v8f acc[4] = {v8f{}, v8f{}, v8f{}, v8f{}};
  for (int k0 = 0; k0 < 768; k0 += 32) {
    const v16bf b = load_b_f32(W, k0, n0, 64, lane);
#pragma unroll
    for (int i = 0; i < 4; ++i) {
      v16bf a = load_a_f32(A, m0 + 16 * i, k0, 768, lane);
      acc[i] = WMMA_BF16(a, b, acc[i]);
    }
  }
  const int n = n0 + (lane & 15);
  const float bn = bias[n];
#pragma unroll
  for (int i = 0; i < 4; ++i) {
    const int mb = m0 + 16 * i + ((lane >> 4) << 3);
#pragma unroll
    for (int j = 0; j < 8; ++j)
      Out[(size_t)(mb + j) * 64 + n] = (__bf16)(acc[i][j] + bn);
  }
}

// ---------------------------------------------------------------------------
// Kernel 2: Wc_eff[d][j] = sum_h Wc[h*64+d][j]  (tile-concat collapse)
// ---------------------------------------------------------------------------
__global__ void wc_reduce_kernel(const float* __restrict__ Wc, __bf16* __restrict__ WcEff) {
  const int t = blockIdx.x * blockDim.x + threadIdx.x;
  if (t >= 64 * 768) return;
  const int d = t / 768, j = t % 768;
  float s = 0.f;
#pragma unroll
  for (int h = 0; h < 12; ++h) s += Wc[(size_t)(h * 64 + d) * 768 + j];
  WcEff[t] = (__bf16)s;
}

// ---------------------------------------------------------------------------
// Kernel 3: scores = (qh @ kh^T)/sqrt(12); softmax; bf16 weights to ws +
// f32 broadcast to 12 heads of d_out as coalesced non-temporal b128 stores.
// grid = (64 m-tiles, 4 batches), block = 256 (8 waves), 64KB LDS
// ---------------------------------------------------------------------------
__global__ __launch_bounds__(256) void scores_softmax_kernel(
    const __bf16* __restrict__ qh, const __bf16* __restrict__ kh,
    __bf16* __restrict__ wbf, float* __restrict__ attn) {
  __shared__ float smem[16 * 1024];
  __shared__ float rowinv[16];
  const int tid  = threadIdx.x;
  const int lane = tid & 31;
  const int wave = tid >> 5;             // 0..7
  const int mt = blockIdx.x;             // 0..63
  const int b  = blockIdx.y;             // 0..3
  const __bf16* Q = qh + (size_t)b * 1024 * 64;
  const __bf16* K = kh + (size_t)b * 1024 * 64;
  const int m0 = mt * 16;
  const float scale = 0.28867513459481287f;  // reference bug: 1/sqrt(H=12)

  // A fragments depend only on the query tile: load once per wave.
  const v16bf a0 = load_a_bf16(Q, m0, 0,  64, lane);
  const v16bf a1 = load_a_bf16(Q, m0, 32, 64, lane);

  // ---- QK^T score tiles into LDS: each wave covers 8 of 64 key tiles ----
#pragma unroll
  for (int t = 0; t < 8; ++t) {
    const int ntile = wave + t * 8;      // 0..63
    v8f acc = {};
    acc = WMMA_BF16(a0, load_bT_bf16(K, ntile * 16, 0,  64, lane), acc);
    acc = WMMA_BF16(a1, load_bT_bf16(K, ntile * 16, 32, 64, lane), acc);
    const int col = ntile * 16 + (lane & 15);
    const int rb  = (lane >> 4) << 3;
#pragma unroll
    for (int j = 0; j < 8; ++j)
      smem[(rb + j) * 1024 + col] = acc[j] * scale;
  }
  __syncthreads();

  // ---- softmax: 16 threads per row (half-wave), shfl_xor reductions ----
  const int row = tid >> 4;   // 0..15
  const int c0  = tid & 15;
  float* srow = smem + row * 1024;

  float mx = -3.4e38f;
  for (int i = c0; i < 1024; i += 16) mx = fmaxf(mx, srow[i]);
#pragma unroll
  for (int off = 8; off; off >>= 1) mx = fmaxf(mx, __shfl_xor(mx, off, 16));

  float sum = 0.f;
  for (int i = c0; i < 1024; i += 16) {
    const float e = __expf(srow[i] - mx);
    srow[i] = e;
    sum += e;
  }
#pragma unroll
  for (int off = 8; off; off >>= 1) sum += __shfl_xor(sum, off, 16);
  if (c0 == 0) rowinv[row] = 1.0f / sum;
  __syncthreads();

  // ---- coalesced float4 write sweep: 16x1024 block, 12-head broadcast ----
  for (int idx4 = tid; idx4 < 4096; idx4 += 256) {
    const int r   = idx4 >> 8;            // 0..15
    const int col = (idx4 & 255) * 4;     // 0..1020, 16B aligned
    f32x4 w = *(const f32x4*)(smem + r * 1024 + col);
    w *= rowinv[r];
    bf16x4 wb;
#pragma unroll
    for (int e = 0; e < 4; ++e) wb[e] = (__bf16)w[e];
    const int gm = m0 + r;
    *(bf16x4*)(wbf + ((size_t)b * 1024 + gm) * 1024 + col) = wb;
#pragma unroll
    for (int h = 0; h < 12; ++h)
      __builtin_nontemporal_store(
          w, (f32x4*)(attn + (((size_t)(b * 12 + h) * 1024 + gm) * 1024 + col)));
  }
}

// ---------------------------------------------------------------------------
// Kernel 4: oh = weights @ vh  ([1024x1024] @ [1024x64] per batch) -> bf16
// Wave computes 64x16 strip: scattered vh B-frag shared by 4 WMMAs per K-step.
// grid = (16 m-quads, 4 n-tiles, 4 batches), block = 32
// ---------------------------------------------------------------------------
__global__ __launch_bounds__(32) void av_kernel(const __bf16* __restrict__ wbf,
                                                const __bf16* __restrict__ vh,
                                                __bf16* __restrict__ oh) {
  const int lane = threadIdx.x & 31;
  const int b = blockIdx.z;
  const __bf16* Wm = wbf + (size_t)b * 1024 * 1024;
  const __bf16* V  = vh  + (size_t)b * 1024 * 64;
  const int m0 = blockIdx.x * 64;
  const int n0 = blockIdx.y * 16;
  v8f acc[4] = {v8f{}, v8f{}, v8f{}, v8f{}};
  for (int k0 = 0; k0 < 1024; k0 += 32) {
    const v16bf bf = load_b_bf16(V, k0, n0, 64, lane);
#pragma unroll
    for (int i = 0; i < 4; ++i) {
      v16bf a = load_a_bf16(Wm, m0 + 16 * i, k0, 1024, lane);
      acc[i] = WMMA_BF16(a, bf, acc[i]);
    }
  }
  const int n = n0 + (lane & 15);
  __bf16* O = oh + (size_t)b * 1024 * 64;
#pragma unroll
  for (int i = 0; i < 4; ++i) {
    const int mb = m0 + 16 * i + ((lane >> 4) << 3);
#pragma unroll
    for (int j = 0; j < 8; ++j)
      O[(size_t)(mb + j) * 64 + n] = (__bf16)acc[i][j];
  }
}

// ---------------------------------------------------------------------------
// Kernel 5: out = oh @ Wc_eff + bc  ([4096x64] @ [64x768]) -> f32 d_out
// Wave computes 64x16 strip; grid = (64 m-quads, 48 n-tiles), block = 32
// ---------------------------------------------------------------------------
__global__ __launch_bounds__(32) void out_kernel(const __bf16* __restrict__ oh,
                                                 const __bf16* __restrict__ WcEff,
                                                 const float* __restrict__ bc,
                                                 float* __restrict__ out) {
  const int lane = threadIdx.x & 31;
  const int m0 = blockIdx.x * 64;
  const int n0 = blockIdx.y * 16;
  v8f acc[4] = {v8f{}, v8f{}, v8f{}, v8f{}};
#pragma unroll
  for (int k0 = 0; k0 < 64; k0 += 32) {
    const v16bf b = load_b_bf16(WcEff, k0, n0, 768, lane);
#pragma unroll
    for (int i = 0; i < 4; ++i) {
      v16bf a = load_a_bf16(oh, m0 + 16 * i, k0, 64, lane);
      acc[i] = WMMA_BF16(a, b, acc[i]);
    }
  }
  const int n = n0 + (lane & 15);
  const float bn = bc[n];
#pragma unroll
  for (int i = 0; i < 4; ++i) {
    const int mb = m0 + 16 * i + ((lane >> 4) << 3);
#pragma unroll
    for (int j = 0; j < 8; ++j)
      out[(size_t)(mb + j) * 768 + n] = acc[i][j] + bn;
  }
}

// ---------------------------------------------------------------------------
extern "C" void kernel_launch(void* const* d_in, const int* in_sizes, int n_in,
                              void* d_out, int out_size, void* d_ws, size_t ws_size,
                              hipStream_t stream) {
  (void)in_sizes; (void)n_in; (void)out_size; (void)ws_size;
  const float* q  = (const float*)d_in[0];
  const float* k  = (const float*)d_in[1];
  const float* v  = (const float*)d_in[2];
  const float* Wq = (const float*)d_in[3];
  const float* bq = (const float*)d_in[4];
  const float* Wk = (const float*)d_in[5];
  const float* bk = (const float*)d_in[6];
  const float* Wv = (const float*)d_in[7];
  const float* bv = (const float*)d_in[8];
  const float* Wc = (const float*)d_in[9];
  const float* bc = (const float*)d_in[10];

  float* out  = (float*)d_out;                         // [4,1024,768]
  float* attn = out + (size_t)4 * 1024 * 768;          // [4,12,1024,1024]

  char* ws = (char*)d_ws;
  size_t off = 0;
  __bf16* qh  = (__bf16*)(ws + off); off += (size_t)4096 * 64 * 2;
  __bf16* kh  = (__bf16*)(ws + off); off += (size_t)4096 * 64 * 2;
  __bf16* vh  = (__bf16*)(ws + off); off += (size_t)4096 * 64 * 2;
  __bf16* oh  = (__bf16*)(ws + off); off += (size_t)4096 * 64 * 2;
  __bf16* wce = (__bf16*)(ws + off); off += (size_t)64 * 768 * 2;
  __bf16* wbf = (__bf16*)(ws + off); // 4*1024*1024*2 = 8 MB

  proj_kernel<<<dim3(64, 4, 3), 32, 0, stream>>>(q, k, v, Wq, Wk, Wv, bq, bk, bv, qh, kh, vh);
  wc_reduce_kernel<<<(64 * 768 + 255) / 256, 256, 0, stream>>>(Wc, wce);
  scores_softmax_kernel<<<dim3(64, 4), 256, 0, stream>>>(qh, kh, wbf, attn);
  av_kernel<<<dim3(16, 4, 4), 32, 0, stream>>>(wbf, vh, oh);
  out_kernel<<<dim3(64, 48), 32, 0, stream>>>(oh, wce, bc, out);
}
